// NeuSModel_49271864819967
// MI455X (gfx1250) — matrix operand order repaired and checked by reading, not compiled, not run on hardware
//
#include <hip/hip_runtime.h>
#include <hip/hip_bf16.h>
#include <math.h>

typedef _Float16 v16h __attribute__((ext_vector_type(16)));
typedef float    v8f  __attribute__((ext_vector_type(8)));

#define HID 64

// ---------------------------------------------------------------------------
// Wave-local LDS fence: same-wave LDS ops are processed in order (DScnt),
// so cross-lane producer->consumer inside one wave only needs the compiler
// not to reorder plus a dscnt drain. Emits the CDNA5 split-counter wait.
// ---------------------------------------------------------------------------
__device__ __forceinline__ void wave_lds_fence() {
    asm volatile("s_wait_dscnt 0x0" ::: "memory");
}

// ---------------------------------------------------------------------------
// WMMA fragment helpers (CDNA5 wave32; A/B use contiguous k order, C/D per CDNA5 ISA 7.12.2)
// ---------------------------------------------------------------------------

// 16-bit A fragment, 16x32 tile from row-major LDS [16][ldm] at column k0.
// lanes 0-15: row M=lane, K = k0 + 0..15
// lanes 16-31: row M=lane-16, K = k0 + 16..31
__device__ __forceinline__ v16h load_a_frag(const _Float16* A, int ldm, int k0) {
    int lane = threadIdx.x & 31;
    int hg = lane >> 4, m = lane & 15;
    const _Float16* p = A + m * ldm + k0 + 16 * hg;
    v16h a = {};
#pragma unroll
    for (int i = 0; i < 16; ++i) a[i] = p[i];
    return a;
}

// 16-bit B fragment, 32x16 tile. B is stored TRANSPOSED in LDS: BT[n][k]
// (stride ldk) so each lane reads contiguous runs (2x ds_load_b128).
// lane = column N (lane&15); lanes 0-15 hold K = k0+0..15, lanes 16-31 K = k0+16..31.
__device__ __forceinline__ v16h load_b_frag(const _Float16* BT, int ldk, int k0, int n0) {
    int lane = threadIdx.x & 31;
    int hg = lane >> 4, n = lane & 15;
    const _Float16* p = BT + (n0 + n) * ldk + k0 + 16 * hg;
    v16h b = {};
#pragma unroll
    for (int i = 0; i < 16; ++i) b[i] = p[i];
    return b;
}

// f32 C/D 16x16: N = lane&15 (+n0), M = r + 8*(lane>>4). Store bias+ReLU as f16.
__device__ __forceinline__ void store_d_relu(_Float16* H, int ldm, int n0, v8f c,
                                             const float* bias) {
    int lane = threadIdx.x & 31;
    int hg = lane >> 4, n = lane & 15;
    float bv = bias[n0 + n];
#pragma unroll
    for (int r = 0; r < 8; ++r) {
        float v = c[r] + bv;
        H[(r + 8 * hg) * ldm + n0 + n] = (_Float16)(v > 0.f ? v : 0.f);
    }
}

__device__ __forceinline__ float sigmoidf(float x) { return 1.f / (1.f + __expf(-x)); }

// ---------------------------------------------------------------------------
// Kernel 1: fused per-sample MLP (forward + analytic SDF gradient + RGB head).
// 128 threads = 4 waves. Each wave runs an INDEPENDENT pipeline over its own
// 16-sample tiles: no block barriers in the steady state, only wave-local
// s_wait_dscnt fences. Weights are staged once to LDS (f16, B transposed).
// ---------------------------------------------------------------------------
__global__ __launch_bounds__(128) void neus_mlp_kernel(
    const float* __restrict__ rays_o, const float* __restrict__ rays_d,
    const float* __restrict__ t_starts, const float* __restrict__ t_ends,
    const int* __restrict__ ray_indices,
    const float* __restrict__ W1, const float* __restrict__ b1,
    const float* __restrict__ W2, const float* __restrict__ b2,
    const float* __restrict__ W3, const float* __restrict__ b3,
    const float* __restrict__ Wr1, const float* __restrict__ br1,
    const float* __restrict__ Wr2, const float* __restrict__ br2,
    const float* __restrict__ s_var,
    float* __restrict__ alpha_o, float* __restrict__ mid_o,
    float* __restrict__ rgb_o, float* __restrict__ nrm_o,
    int M, int numTiles16)
{
    // ---- weights in LDS, f16, B matrices stored transposed ([N][K]) ----
    __shared__ __align__(16) _Float16 sW1f[64][32];  // fwd L1: B=W1  (Kpad32 x 64)
    __shared__ __align__(16) _Float16 sW2f[64][64];  // fwd L2: B=W2
    __shared__ __align__(16) _Float16 sW2b[64][64];  // bwd   : B=W2^T -> store W2 row-major
    __shared__ __align__(16) _Float16 sW3f[32][64];  // fwd L3: B=W3  (64 x Npad32)
    __shared__ __align__(16) _Float16 sW1b[16][64];  // bwd   : B=W1^T (64 x Npad16)
    __shared__ __align__(16) _Float16 sWr1[64][32];  // rgb L1: B=Wr1 (Kpad32 x 64)
    __shared__ __align__(16) _Float16 sWr2[16][64];  // rgb L2: B=Wr2 (64 x Npad16)
    __shared__ float sb1[64], sb2[64], sb3[32], sbr1[64], sbr2[4], sw3c0[64];
    // ---- per-wave tiles (wave-private) ----
    __shared__ __align__(16) _Float16 sH1[4][16][HID];
    __shared__ __align__(16) _Float16 sH2[4][16][HID];
    __shared__ __align__(16) _Float16 sT [4][16][HID];   // input / d2 / rgb-in scratch
    __shared__ __align__(16) _Float16 sFeat[4][16][16];
    __shared__ __align__(16) _Float16 sPD[4][16][8];     // pos[0..2], dir[3..5]
    __shared__ float sSdf[4][16];
    __shared__ float sGrad[4][16][4];
    __shared__ float sRgb[4][16][4];

    const int t = threadIdx.x;
    const int wave = t >> 5;
    const int lane = t & 31;

    // ---- one-time weight staging (f32 global -> f16 LDS) ----
    for (int i = t; i < 64 * 32; i += 128) { int n = i >> 5, k = i & 31;
        sW1f[n][k] = (_Float16)((k < 3) ? W1[k * HID + n] : 0.f);
        sWr1[n][k] = (_Float16)((k < 25) ? Wr1[k * HID + n] : 0.f); }
    for (int i = t; i < 64 * 64; i += 128) { int n = i >> 6, k = i & 63;
        sW2f[n][k] = (_Float16)W2[k * HID + n];
        sW2b[n][k] = (_Float16)W2[n * HID + k]; }
    for (int i = t; i < 32 * 64; i += 128) { int n = i >> 6, k = i & 63;
        sW3f[n][k] = (_Float16)((n < 17) ? W3[k * 17 + n] : 0.f); }
    for (int i = t; i < 16 * 64; i += 128) { int n = i >> 6, k = i & 63;
        sW1b[n][k] = (_Float16)((n < 3) ? W1[n * HID + k] : 0.f);
        sWr2[n][k] = (_Float16)((n < 3) ? Wr2[k * 3 + n] : 0.f); }
    if (t < 64) { sb1[t] = b1[t]; sb2[t] = b2[t]; sbr1[t] = br1[t]; sw3c0[t] = W3[t * 17]; }
    if (t < 32) sb3[t] = (t < 17) ? b3[t] : 0.f;
    if (t < 4)  sbr2[t] = (t < 3) ? br2[t] : 0.f;
    const float inv_s = fminf(fmaxf(__expf(s_var[0]), 1e-6f), 1e6f);
    __syncthreads();   // only block-wide barrier: weights are read-only below

    const int waveStride = gridDim.x * 4;
    for (int tile = blockIdx.x * 4 + wave; tile < numTiles16; tile += waveStride) {
        const int s0 = tile * 16;

        // ---- prefetch next tile's per-sample scalars (global_prefetch_b8) ----
        {
            int sn = s0 + waveStride * 16 + (lane & 15);
            if (lane < 16 && sn < M) {
                __builtin_prefetch(&ray_indices[sn], 0, 0);
                __builtin_prefetch(&t_starts[sn], 0, 0);
                __builtin_prefetch(&t_ends[sn], 0, 0);
            }
        }

        // ---- build input tile (wave-private; lanes 0-15 real work, 16-31 pad) ----
        {
            int m = lane & 15;
            if (lane < 16) {
                int s = s0 + m;
                float px = 0, py = 0, pz = 0, dx = 0, dy = 0, dz = 0;
                if (s < M) {
                    int r = ray_indices[s];
                    float ox = rays_o[3 * r], oy = rays_o[3 * r + 1], oz = rays_o[3 * r + 2];
                    float rx = rays_d[3 * r], ry = rays_d[3 * r + 1], rz = rays_d[3 * r + 2];
                    float inv = 1.f / fmaxf(sqrtf(rx * rx + ry * ry + rz * rz), 1e-12f);
                    dx = rx * inv; dy = ry * inv; dz = rz * inv;
                    float mid = 0.5f * (t_starts[s] + t_ends[s]);
                    px = ox + dx * mid; py = oy + dy * mid; pz = oz + dz * mid;
                }
                sT[wave][m][0] = (_Float16)px;
                sT[wave][m][1] = (_Float16)py;
                sT[wave][m][2] = (_Float16)pz;
#pragma unroll
                for (int k = 3; k < 16; ++k) sT[wave][m][k] = (_Float16)0.f;
                sPD[wave][m][0] = (_Float16)px; sPD[wave][m][1] = (_Float16)py;
                sPD[wave][m][2] = (_Float16)pz; sPD[wave][m][3] = (_Float16)dx;
                sPD[wave][m][4] = (_Float16)dy; sPD[wave][m][5] = (_Float16)dz;
            } else {
#pragma unroll
                for (int k = 16; k < 32; ++k) sT[wave][m][k] = (_Float16)0.f;
            }
        }
        wave_lds_fence();

        // ---- L1: h1 = relu(pos @ W1 + b1)  [16x32]x[32x64] ----
#pragma unroll
        for (int nt = 0; nt < 4; ++nt) {
            v16h a = load_a_frag(&sT[wave][0][0], HID, 0);
            v16h b = load_b_frag(&sW1f[0][0], 32, 0, nt * 16);
            v8f c = {};
            c = __builtin_amdgcn_wmma_f32_16x16x32_f16(false, a, false, b, (short)0, c, false, false);
            store_d_relu(&sH1[wave][0][0], HID, nt * 16, c, sb1);
        }
        wave_lds_fence();

        // ---- L2: h2 = relu(h1 @ W2 + b2)  [16x64]x[64x64] ----
#pragma unroll
        for (int nt = 0; nt < 4; ++nt) {
            v8f c = {};
#pragma unroll
            for (int kc = 0; kc < 2; ++kc) {
                v16h a = load_a_frag(&sH1[wave][0][0], HID, kc * 32);
                v16h b = load_b_frag(&sW2f[0][0], HID, kc * 32, nt * 16);
                c = __builtin_amdgcn_wmma_f32_16x16x32_f16(false, a, false, b, (short)0, c, false, false);
            }
            store_d_relu(&sH2[wave][0][0], HID, nt * 16, c, sb2);
        }
        wave_lds_fence();

        // ---- L3: o = h2 @ W3 + b3 -> sdf (n=0), feat (n=1..16) ----
#pragma unroll
        for (int nt = 0; nt < 2; ++nt) {
            v8f c = {};
#pragma unroll
            for (int kc = 0; kc < 2; ++kc) {
                v16h a = load_a_frag(&sH2[wave][0][0], HID, kc * 32);
                v16h b = load_b_frag(&sW3f[0][0], HID, kc * 32, nt * 16);
                c = __builtin_amdgcn_wmma_f32_16x16x32_f16(false, a, false, b, (short)0, c, false, false);
            }
            int hg = lane >> 4, n = lane & 15, ng = nt * 16 + n;
            float bv = sb3[ng];
#pragma unroll
            for (int r = 0; r < 8; ++r) {
                int m = r + 8 * hg;
                float v = c[r] + bv;
                if (ng == 0) sSdf[wave][m] = v;
                else if (ng < 17) sFeat[wave][m][ng - 1] = (_Float16)v;
            }
        }
        wave_lds_fence();

        // ---- backward: d2 = relu'(z2) * W3[:,0] ----
        for (int i = lane; i < 16 * HID; i += 32) {
            int m = i >> 6, k = i & 63;
            float h = (float)sH2[wave][m][k];
            sT[wave][m][k] = (_Float16)((h > 0.f) ? sw3c0[k] : 0.f);
        }
        wave_lds_fence();

        // ---- d1 = relu'(z1) * (d2 @ W2^T) ----
#pragma unroll
        for (int nt = 0; nt < 4; ++nt) {
            v8f c = {};
#pragma unroll
            for (int kc = 0; kc < 2; ++kc) {
                v16h a = load_a_frag(&sT[wave][0][0], HID, kc * 32);
                v16h b = load_b_frag(&sW2b[0][0], HID, kc * 32, nt * 16);
                c = __builtin_amdgcn_wmma_f32_16x16x32_f16(false, a, false, b, (short)0, c, false, false);
            }
            int hg = lane >> 4, n = lane & 15, ng = nt * 16 + n;
#pragma unroll
            for (int r = 0; r < 8; ++r) {
                int m = r + 8 * hg;
                float h1v = (float)sH1[wave][m][ng];
                sH2[wave][m][ng] = (_Float16)((h1v > 0.f) ? c[r] : 0.f);
            }
        }
        wave_lds_fence();

        // ---- grad = d1 @ W1^T (3 cols) ----
        {
            v8f c = {};
#pragma unroll
            for (int kc = 0; kc < 2; ++kc) {
                v16h a = load_a_frag(&sH2[wave][0][0], HID, kc * 32);
                v16h b = load_b_frag(&sW1b[0][0], HID, kc * 32, 0);
                c = __builtin_amdgcn_wmma_f32_16x16x32_f16(false, a, false, b, (short)0, c, false, false);
            }
            int hg = lane >> 4, n = lane & 15;
            if (n < 3) {
#pragma unroll
                for (int r = 0; r < 8; ++r) sGrad[wave][r + 8 * hg][n] = c[r];
            }
        }
        wave_lds_fence();

        // ---- rgb input tile: [pos, normal, dir, feat] padded to 32 ----
        for (int i = lane; i < 16 * 32; i += 32) {
            int m = i >> 5, k = i & 31;
            float v;
            if (k < 3) v = (float)sPD[wave][m][k];
            else if (k < 6) {
                float gx = sGrad[wave][m][0], gy = sGrad[wave][m][1], gz = sGrad[wave][m][2];
                float inv = 1.f / fmaxf(sqrtf(gx * gx + gy * gy + gz * gz), 1e-12f);
                v = ((k == 3) ? gx : (k == 4) ? gy : gz) * inv;
            }
            else if (k < 9) v = (float)sPD[wave][m][k - 3];
            else if (k < 25) v = (float)sFeat[wave][m][k - 9];
            else v = 0.f;
            sT[wave][m][k] = (_Float16)v;
        }
        wave_lds_fence();

        // ---- rgb L1: hr = relu(rgb_in @ Wr1 + br1) ----
#pragma unroll
        for (int nt = 0; nt < 4; ++nt) {
            v16h a = load_a_frag(&sT[wave][0][0], HID, 0);
            v16h b = load_b_frag(&sWr1[0][0], 32, 0, nt * 16);
            v8f c = {};
            c = __builtin_amdgcn_wmma_f32_16x16x32_f16(false, a, false, b, (short)0, c, false, false);
            store_d_relu(&sH1[wave][0][0], HID, nt * 16, c, sbr1);
        }
        wave_lds_fence();

        // ---- rgb L2: hr @ Wr2 + br2 (3 cols) ----
        {
            v8f c = {};
#pragma unroll
            for (int kc = 0; kc < 2; ++kc) {
                v16h a = load_a_frag(&sH1[wave][0][0], HID, kc * 32);
                v16h b = load_b_frag(&sWr2[0][0], HID, kc * 32, 0);
                c = __builtin_amdgcn_wmma_f32_16x16x32_f16(false, a, false, b, (short)0, c, false, false);
            }
            int hg = lane >> 4, n = lane & 15;
            if (n < 3) {
#pragma unroll
                for (int r = 0; r < 8; ++r) sRgb[wave][r + 8 * hg][n] = c[r] + sbr2[n];
            }
        }
        wave_lds_fence();

        // ---- per-sample epilogue: NeuS alpha + per-sample buffers (lanes 0-15) ----
        if (lane < 16) {
            int m = lane, s = s0 + m;
            if (s < M) {
                float ts = t_starts[s], te = t_ends[s];
                float mid = 0.5f * (ts + te), dist = te - ts;
                float sdf = sSdf[wave][m];
                float prev = sigmoidf((sdf + dist * 0.5f) * inv_s);
                float next = sigmoidf((sdf - dist * 0.5f) * inv_s);
                float alpha = (prev - next) / (prev + 1e-5f);
                alpha = fminf(fmaxf(alpha, 0.f), 1.f);
                alpha_o[s] = alpha;
                mid_o[s] = mid;
                rgb_o[3 * s + 0] = sigmoidf(sRgb[wave][m][0]);
                rgb_o[3 * s + 1] = sigmoidf(sRgb[wave][m][1]);
                rgb_o[3 * s + 2] = sigmoidf(sRgb[wave][m][2]);
                float gx = sGrad[wave][m][0], gy = sGrad[wave][m][1], gz = sGrad[wave][m][2];
                float inv = 1.f / fmaxf(sqrtf(gx * gx + gy * gy + gz * gz), 1e-12f);
                nrm_o[3 * s + 0] = gx * inv;
                nrm_o[3 * s + 1] = gy * inv;
                nrm_o[3 * s + 2] = gz * inv;
            }
        }
        wave_lds_fence();   // tile buffers free for next iteration
    }
}

// ---------------------------------------------------------------------------
// Kernel 2: deterministic per-ray compositing (ray_indices is sorted).
// One thread per ray: binary-search segment bounds, serial transmittance walk.
// out layout: [rgb 3N | depth N | opacity N | normal 3N]
// ---------------------------------------------------------------------------
__global__ __launch_bounds__(256) void neus_render_kernel(
    const int* __restrict__ ray, const float* __restrict__ alpha,
    const float* __restrict__ mid, const float* __restrict__ rgb,
    const float* __restrict__ nrm, float* __restrict__ out, int M, int N)
{
    int r = blockIdx.x * blockDim.x + threadIdx.x;
    if (r >= N) return;
    int lo = 0, hi = M;
    while (lo < hi) { int md = (lo + hi) >> 1; if (ray[md] < r) lo = md + 1; else hi = md; }
    int s = lo;
    hi = M;
    while (lo < hi) { int md = (lo + hi) >> 1; if (ray[md] <= r) lo = md + 1; else hi = md; }
    int e = lo;

    float T = 1.f, op = 0.f, dp = 0.f;
    float cr = 0.f, cg = 0.f, cb = 0.f, nx = 0.f, ny = 0.f, nz = 0.f;
    for (int i = s; i < e; ++i) {
        float a = alpha[i];
        float w = a * T;
        op += w; dp += w * mid[i];
        cr += w * rgb[3 * i];     cg += w * rgb[3 * i + 1]; cb += w * rgb[3 * i + 2];
        nx += w * nrm[3 * i];     ny += w * nrm[3 * i + 1]; nz += w * nrm[3 * i + 2];
        T *= fminf(fmaxf(1.f - a, 1e-10f), 1.f);
    }
    float inv = 1.f / fmaxf(sqrtf(nx * nx + ny * ny + nz * nz), 1e-12f);
    out[3 * r + 0] = cr; out[3 * r + 1] = cg; out[3 * r + 2] = cb;
    out[3 * N + r] = dp;
    out[4 * N + r] = op;
    out[5 * N + 3 * r + 0] = nx * inv;
    out[5 * N + 3 * r + 1] = ny * inv;
    out[5 * N + 3 * r + 2] = nz * inv;
}

// ---------------------------------------------------------------------------
extern "C" void kernel_launch(void* const* d_in, const int* in_sizes, int n_in,
                              void* d_out, int out_size, void* d_ws, size_t ws_size,
                              hipStream_t stream) {
    const float* rays_o      = (const float*)d_in[0];
    const float* rays_d      = (const float*)d_in[1];
    const float* t_starts    = (const float*)d_in[2];
    const float* t_ends      = (const float*)d_in[3];
    const int*   ray_indices = (const int*)  d_in[4];
    const float* W1  = (const float*)d_in[5];
    const float* b1  = (const float*)d_in[6];
    const float* W2  = (const float*)d_in[7];
    const float* b2  = (const float*)d_in[8];
    const float* W3  = (const float*)d_in[9];
    const float* b3  = (const float*)d_in[10];
    const float* Wr1 = (const float*)d_in[11];
    const float* br1 = (const float*)d_in[12];
    const float* Wr2 = (const float*)d_in[13];
    const float* br2 = (const float*)d_in[14];
    const float* sv  = (const float*)d_in[15];

    const int M = in_sizes[2];       // t_starts: M samples
    const int N = in_sizes[0] / 3;   // rays_o: N rays

    // workspace (floats): alpha[M] mid[M] rgb[3M] nrm[3M]
    float* ws    = (float*)d_ws;
    float* alpha = ws;
    float* mid   = ws + (size_t)M;
    float* rgb   = ws + (size_t)2 * M;
    float* nrm   = ws + (size_t)5 * M;

    const int numTiles16 = (M + 15) / 16;          // one 16-sample tile per wave
    int blocks = (numTiles16 + 3) / 4;             // 4 waves per block
    if (blocks > 4096) blocks = 4096;

    neus_mlp_kernel<<<blocks, 128, 0, stream>>>(
        rays_o, rays_d, t_starts, t_ends, ray_indices,
        W1, b1, W2, b2, W3, b3, Wr1, br1, Wr2, br2, sv,
        alpha, mid, rgb, nrm, M, numTiles16);

    neus_render_kernel<<<(N + 255) / 256, 256, 0, stream>>>(
        ray_indices, alpha, mid, rgb, nrm, (float*)d_out, M, N);
}
